// MyRNN_38113539785460
// MI455X (gfx1250) — matrix operand order repaired and checked by reading, not compiled, not run on hardware
//
#include <hip/hip_runtime.h>

typedef __attribute__((ext_vector_type(16))) _Float16 v16h;
typedef __attribute__((ext_vector_type(8)))  float    v8f;

#define Hn    64
#define Tn    86
#define Bn    4096
#define Vn    257
#define ROWS  16          // batch rows per workgroup
#define XS    72          // LDS half-stride for X/H rows (bank-spread)
#define GS    260         // LDS float-stride for gates rows (kills 2-way conflict)

union FragH {
    v16h     v;
    _Float16 h[16];
    unsigned u[8];
};

// ---- A fragment (16x32 f16, M x K) from LDS, per ISA 7.12.2 ----
// lane<16 : row M=lane, VGPR j<4 -> K=kb+2j,2j+1 ; j>=4 -> K=kb+8+2j (i.e. 16..23)
// lane>=16: same rows, K offsets +8 (8..15 / 24..31)
__device__ inline v16h loadA(const _Float16* base, int lane, int kb) {
    FragH f;
    const int m  = lane & 15;
    const int ko = kb + ((lane & 16) ? 8 : 0);
    const _Float16* p = base + m * XS + ko;
#pragma unroll
    for (int j = 0; j < 8; ++j) {
        const int off = (j < 4) ? (2 * j) : (8 + 2 * j);
        f.u[j] = *(const unsigned*)(p + off);
    }
    return f.v;
}

// ---- B fragment (32x16 f16, K x N) built from GLOBAL f32 weights (row-major [rows][64]) ----
// B[k][n] = W[row0+n][kb+k]; lanes 0-15: K=kb+0..7,16..23, lanes 16-31: K=kb+8..15,24..31.
// Rows beyond `rows` (vocab padding) read as zero.
__device__ inline v16h loadB_g(const float* __restrict__ w, int rows, int lane,
                               int row0, int kb) {
    FragH f;
    const int n  = lane & 15;
    const int r  = row0 + n;
    const int ko = kb + ((lane & 16) ? 8 : 0);
#pragma unroll
    for (int e = 0; e < 16; ++e) {
        const float val = (r < rows) ? w[r * Hn + ko + ((e < 8) ? e : (e + 8))] : 0.0f;
        f.h[e] = (_Float16)val;
    }
    return f.v;
}

// Branch-free transcendentals on the hardware TRANS pipe (v_exp_f32 / v_rcp_f32).
#define LOG2E  1.44269504088896f
__device__ inline float fast_sigmoid(float x) {
    return __builtin_amdgcn_rcpf(1.0f + __builtin_amdgcn_exp2f(-x * LOG2E));
}
__device__ inline float fast_tanh(float x) {
    return 1.0f - 2.0f * __builtin_amdgcn_rcpf(1.0f + __builtin_amdgcn_exp2f(x * (2.0f * LOG2E)));
}

__global__ __launch_bounds__(256) void lstm_lm_fused(
    const int*   __restrict__ idx,  const float* __restrict__ wte,
    const float* __restrict__ W_ih, const float* __restrict__ W_hh,
    const float* __restrict__ b_ih, const float* __restrict__ b_hh,
    const float* __restrict__ lm_W, const float* __restrict__ lm_b,
    float* __restrict__ out)
{
    __shared__ _Float16 sX[ROWS * XS];     // current embeddings (f16)
    __shared__ _Float16 sH[ROWS * XS];     // current hidden state (f16)
    __shared__ float    sG[ROWS * GS];     // gate pre-activations (f32)
    __shared__ float    sBias[256];        // b_ih + b_hh
    __shared__ float    sLmb[17 * 16];     // lm_b padded to 272

    const int tid  = threadIdx.x;
    const int lane = tid & 31;
    const int wv   = tid >> 5;             // wave id 0..7
    const int r0   = blockIdx.x * ROWS;    // first batch row

    // ---- init small LDS state ----
    if (tid < 256) sBias[tid] = b_ih[tid] + b_hh[tid];
    for (int i = tid; i < 17 * 16; i += 256) sLmb[i] = (i < Vn) ? lm_b[i] : 0.0f;
    for (int i = tid; i < ROWS * XS; i += 256) sH[i] = (_Float16)0.0f;

    // ---- hoist ALL weight B-fragments into registers (loop-invariant) ----
    const int gt0 = (2 * wv) * 16;         // gate tile pair: columns [32w, 32w+32)
    const int gt1 = (2 * wv + 1) * 16;
    const v16h bI00 = loadB_g(W_ih, 256, lane, gt0, 0);
    const v16h bI01 = loadB_g(W_ih, 256, lane, gt0, 32);
    const v16h bI10 = loadB_g(W_ih, 256, lane, gt1, 0);
    const v16h bI11 = loadB_g(W_ih, 256, lane, gt1, 32);
    const v16h bH00 = loadB_g(W_hh, 256, lane, gt0, 0);
    const v16h bH01 = loadB_g(W_hh, 256, lane, gt0, 32);
    const v16h bH10 = loadB_g(W_hh, 256, lane, gt1, 0);
    const v16h bH11 = loadB_g(W_hh, 256, lane, gt1, 32);
    // logits tiles: wave w handles {2w, 2w+1}; wave 7 additionally tile 16 (vocab pad)
    const v16h bL00 = loadB_g(lm_W, Vn, lane, gt0, 0);
    const v16h bL01 = loadB_g(lm_W, Vn, lane, gt0, 32);
    const v16h bL10 = loadB_g(lm_W, Vn, lane, gt1, 0);
    const v16h bL11 = loadB_g(lm_W, Vn, lane, gt1, 32);
    const v16h bL20 = loadB_g(lm_W, Vn, lane, 16 * 16, 0);
    const v16h bL21 = loadB_g(lm_W, Vn, lane, 16 * 16, 32);

    // cell state lives in registers: this thread owns (row cm, hidden cj..cj+3)
    const int cm = tid & 15;
    const int cj = (tid >> 4) * 4;
    float creg[4] = {0.0f, 0.0f, 0.0f, 0.0f};

    const int nlo = lane & 15;             // N within tile
    const int mb  = (lane & 16) ? 8 : 0;   // C/D: upper half-lanes hold M=r+8

    // ---- embedding gather is software-pipelined: this thread owns
    //      (row em = tid>>4, cols ek0..ek0+3); prefetch t=0 now ----
    const int em  = tid >> 4;
    const int ek0 = (tid & 15) * 4;
    const int* idxRow = idx + (r0 + em) * Tn;
    float4 wpre = *(const float4*)(wte + idxRow[0] * Hn + ek0);

    __syncthreads();

    for (int t = 0; t < Tn; ++t) {
        // ---- 1) commit prefetched embedding for step t ----
        sX[em * XS + ek0 + 0] = (_Float16)wpre.x;
        sX[em * XS + ek0 + 1] = (_Float16)wpre.y;
        sX[em * XS + ek0 + 2] = (_Float16)wpre.z;
        sX[em * XS + ek0 + 3] = (_Float16)wpre.w;
        __syncthreads();

        // ---- 1b) prefetch embedding for step t+1 (hidden under GEMM/cell) ----
        {
            const int tn = (t + 1 < Tn) ? (t + 1) : t;
            wpre = *(const float4*)(wte + idxRow[tn] * Hn + ek0);
        }

        // ---- 2) gate GEMM: gates[16,256] = X@W_ih^T + H@W_hh^T + bias ----
        const v16h ax0 = loadA(sX, lane, 0);
        const v16h ax1 = loadA(sX, lane, 32);
        const v16h ah0 = loadA(sH, lane, 0);
        const v16h ah1 = loadA(sH, lane, 32);
        {
            const float bv = sBias[gt0 + nlo];
            v8f acc = {bv, bv, bv, bv, bv, bv, bv, bv};
            acc = __builtin_amdgcn_wmma_f32_16x16x32_f16(false, ax0, false, bI00, (short)0, acc, false, false);
            acc = __builtin_amdgcn_wmma_f32_16x16x32_f16(false, ax1, false, bI01, (short)0, acc, false, false);
            acc = __builtin_amdgcn_wmma_f32_16x16x32_f16(false, ah0, false, bH00, (short)0, acc, false, false);
            acc = __builtin_amdgcn_wmma_f32_16x16x32_f16(false, ah1, false, bH01, (short)0, acc, false, false);
#pragma unroll
            for (int r = 0; r < 8; ++r) sG[(r + mb) * GS + gt0 + nlo] = acc[r];
        }
        {
            const float bv = sBias[gt1 + nlo];
            v8f acc = {bv, bv, bv, bv, bv, bv, bv, bv};
            acc = __builtin_amdgcn_wmma_f32_16x16x32_f16(false, ax0, false, bI10, (short)0, acc, false, false);
            acc = __builtin_amdgcn_wmma_f32_16x16x32_f16(false, ax1, false, bI11, (short)0, acc, false, false);
            acc = __builtin_amdgcn_wmma_f32_16x16x32_f16(false, ah0, false, bH10, (short)0, acc, false, false);
            acc = __builtin_amdgcn_wmma_f32_16x16x32_f16(false, ah1, false, bH11, (short)0, acc, false, false);
#pragma unroll
            for (int r = 0; r < 8; ++r) sG[(r + mb) * GS + gt1 + nlo] = acc[r];
        }
        __syncthreads();

        // ---- 3) LSTM cell (gate order i,f,g,o), c in registers ----
#pragma unroll
        for (int q = 0; q < 4; ++q) {
            const int j  = cj + q;
            const float gi = sG[cm * GS + j];
            const float gf = sG[cm * GS + 64 + j];
            const float gg = sG[cm * GS + 128 + j];
            const float go = sG[cm * GS + 192 + j];
            const float cn = fast_sigmoid(gf) * creg[q] + fast_sigmoid(gi) * fast_tanh(gg);
            creg[q] = cn;
            const float hh = fast_sigmoid(go) * fast_tanh(cn);
            sH[cm * XS + j] = (_Float16)hh;
        }
        __syncthreads();

        // ---- 4) logits GEMM: out[16, 257] = H @ lm_W^T + lm_b ----
        const v16h lh0 = loadA(sH, lane, 0);
        const v16h lh1 = loadA(sH, lane, 32);
        auto doLogit = [&](int nt, v16h bk0, v16h bk1) {
            const float bv = sLmb[nt * 16 + nlo];
            v8f acc = {bv, bv, bv, bv, bv, bv, bv, bv};
            acc = __builtin_amdgcn_wmma_f32_16x16x32_f16(false, lh0, false, bk0, (short)0, acc, false, false);
            acc = __builtin_amdgcn_wmma_f32_16x16x32_f16(false, lh1, false, bk1, (short)0, acc, false, false);
            const int v = nt * 16 + nlo;
            if (v < Vn) {
#pragma unroll
                for (int r = 0; r < 8; ++r) {
                    const size_t b = (size_t)(r0 + r + mb);
                    out[b * (size_t)(Tn * Vn) + (size_t)t * Vn + v] = acc[r];
                }
            }
        };
        doLogit(2 * wv, bL00, bL01);
        doLogit(2 * wv + 1, bL10, bL11);
        if (wv == 7) doLogit(16, bL20, bL21);
        // next-step sX write is fenced by the post-commit barrier before
        // anyone re-reads sX/sH.
    }
}

extern "C" void kernel_launch(void* const* d_in, const int* in_sizes, int n_in,
                              void* d_out, int out_size, void* d_ws, size_t ws_size,
                              hipStream_t stream) {
    (void)in_sizes; (void)n_in; (void)d_ws; (void)ws_size; (void)out_size;
    const int*   idx  = (const int*)  d_in[0];
    const float* wte  = (const float*)d_in[1];
    const float* W_ih = (const float*)d_in[2];
    const float* W_hh = (const float*)d_in[3];
    const float* b_ih = (const float*)d_in[4];
    const float* b_hh = (const float*)d_in[5];
    const float* lm_W = (const float*)d_in[6];
    const float* lm_b = (const float*)d_in[7];
    float* out = (float*)d_out;

    dim3 grid(Bn / ROWS);   // 256 workgroups, each: 16 batch rows, 8 waves
    lstm_lm_fused<<<grid, 256, 0, stream>>>(idx, wte, W_ih, W_hh, b_ih, b_hh,
                                            lm_W, lm_b, out);
}
